// Custom_Model_Quant_Embedding_Group_69458211111667
// MI455X (gfx1250) — matrix operand order repaired and checked by reading, not compiled, not run on hardware
//
#include <hip/hip_runtime.h>

typedef __attribute__((ext_vector_type(16))) _Float16     v16h;
typedef __attribute__((ext_vector_type(8)))  float        v8f;
typedef __attribute__((ext_vector_type(8)))  unsigned int v8u;

#define BAGS_PER_BLOCK 4
#define MAX_ROWS   64            // covers L=50; two K-chunks of 32
#define ROW_BYTES  64            // 128 int4 cols = 64 packed bytes per row
#define T_STRIDE   68            // transposed-LDS stride (17 dwords, odd -> bank-conflict-free)

__global__ __launch_bounds__(128)
void quant_embed_bag_wmma(const int*   __restrict__ weights,  // [V, 64] int32, low byte = 2 nibbles
                          const float* __restrict__ scales,   // [V]
                          const float* __restrict__ zeros,    // [V]
                          const int*   __restrict__ indices,  // [N]
                          const int*   __restrict__ offsets,  // [B+1]
                          const float* __restrict__ other,    // [B, 128]
                          float*       __restrict__ out,      // [B, 384]
                          int B)
{
    // transposed packed bytes: ldsT[bag][byteCol * T_STRIDE + permRow]
    // rows permuted within groups of 4 (0,1,2,3 -> bytes 0,2,1,3) so that A-slot
    // pairs come out of a loaded dword with one AND+OR each.
    __shared__ unsigned char ldsT[BAGS_PER_BLOCK][ROW_BYTES * T_STRIDE];
    __shared__ __attribute__((aligned(16))) float ldsScale[BAGS_PER_BLOCK][MAX_ROWS];
    __shared__ int ldsIdx[BAGS_PER_BLOCK][MAX_ROWS];

    const int lane = threadIdx.x & 31;
    const int wid  = threadIdx.x >> 5;
    int bag = blockIdx.x * BAGS_PER_BLOCK + wid;
    if (bag >= B) bag = B - 1;            // clamp; keep barriers uniform

    const int beg = offsets[bag];
    int n = offsets[bag + 1] - beg;
    if (n > MAX_ROWS) n = MAX_ROWS;

    unsigned char* rowT = ldsT[wid];
    float*         sBuf = ldsScale[wid];
    int*           iBuf = ldsIdx[wid];

    // ---------- phase 1: idx/scale/zero gather; zero-fill tails ----------
    // A is fed biased codes (1024+q), so fold the bias into the correction:
    // corr = sum_k s_k * (z_k + 1024)
    float corrPart = 0.0f;
    #pragma unroll
    for (int k0 = 0; k0 < MAX_ROWS; k0 += 32) {
        int   k = k0 + lane;
        int   idx = 0;
        float s = 0.0f, z = 0.0f;
        if (k < n) {
            idx = indices[beg + k];
            s = scales[idx];
            z = zeros[idx];
        }
        sBuf[k] = s;
        iBuf[k] = idx;
        corrPart += s * (z + 1024.0f);
    }
    #pragma unroll
    for (int m = 16; m >= 1; m >>= 1)
        corrPart += __shfl_xor(corrPart, m, 32);
    const float corr = corrPart;

    __syncthreads();   // iBuf/sBuf visible across lanes

    // ---------- phase 2: coalesced 256B row gather -> byte-transposed LDS ----------
    // 2 rows per iteration; 16 lanes per row each load int4 (4 int32 = 4 packed bytes)
    const int half = lane >> 4;
    const int sub  = lane & 15;
    for (int it = 0; it * 2 < n; ++it) {
        int r = it * 2 + half;
        if (r < n) {
            int idx = iBuf[r];
            int4 w = *((const int4*)(weights + (size_t)idx * 64) + sub);
            int rp = (r & ~3) | ((r & 1) << 1) | ((r >> 1) & 1); // swap middle-two perm
            unsigned char* col = rowT + (sub * 4) * T_STRIDE + rp;
            col[0 * T_STRIDE] = (unsigned char)w.x;
            col[1 * T_STRIDE] = (unsigned char)w.y;
            col[2 * T_STRIDE] = (unsigned char)w.z;
            col[3 * T_STRIDE] = (unsigned char)w.w;
        }
    }
    __syncthreads();   // rowT visible across lanes

    // ---------- phase 3: B fragments = scales broadcast along N (unconditional) ----------
    // f16 B 32x16: lanes 0-15 hold K=0..7,16..23; lanes 16-31 hold K=8..15,24..31
    const int bK = (lane & 16) ? 8 : 0;
    const float4* sv0 = (const float4*)(sBuf + bK);       // K = bK..bK+7, 16+bK..16+bK+7
    const float4* sv1 = (const float4*)(sBuf + 32 + bK);  // same, +32
    v16h bfrag0, bfrag1;
    #pragma unroll
    for (int j = 0; j < 4; ++j) {
        float4 f0 = sv0[j + (j >> 1) * 2], f1 = sv1[j + (j >> 1) * 2];
        bfrag0[4*j+0] = (_Float16)f0.x;  bfrag0[4*j+1] = (_Float16)f0.y;
        bfrag0[4*j+2] = (_Float16)f0.z;  bfrag0[4*j+3] = (_Float16)f0.w;
        bfrag1[4*j+0] = (_Float16)f1.x;  bfrag1[4*j+1] = (_Float16)f1.y;
        bfrag1[4*j+2] = (_Float16)f1.z;  bfrag1[4*j+3] = (_Float16)f1.w;
    }

    // ---------- phase 4: 8 column tiles x 2 K-chunks of WMMA ----------
    // A 16x32 f16 layout: lane m=lane&15; lanes 0-15: halfs0-7 K=0..7, halfs8-15 K=16..23;
    // lanes 16-31: +8 on both K ranges. Slots carry f16(1024+q) = 0x6400 | q (exact).
    const int m        = lane & 15;
    const int aKbase   = (lane & 16) ? 8 : 0;
    const int byteIdx  = m >> 1;
    const int nibShift = (m & 1) * 4;
    const size_t outBase = (size_t)bag * 384;

    #pragma unroll
    for (int t = 0; t < 8; ++t) {
        const unsigned char* colBase = rowT + (t * 8 + byteIdx) * T_STRIDE + aKbase;
        v8f acc = {};
        #pragma unroll
        for (int ch = 0; ch < 2; ++ch) {
            const unsigned char* p = colBase + ch * 32;
            unsigned int w0 = (*(const unsigned int*)(p +  0)) >> nibShift; // K off 0..3
            unsigned int w1 = (*(const unsigned int*)(p +  4)) >> nibShift; // K off 4..7
            unsigned int w2 = (*(const unsigned int*)(p + 16)) >> nibShift; // K off 16..19
            unsigned int w3 = (*(const unsigned int*)(p + 20)) >> nibShift; // K off 20..23
            v8u au;
            au[0] = ( w0       & 0x000F000Fu) | 0x64006400u;
            au[1] = ((w0 >> 8) & 0x000F000Fu) | 0x64006400u;
            au[2] = ( w1       & 0x000F000Fu) | 0x64006400u;
            au[3] = ((w1 >> 8) & 0x000F000Fu) | 0x64006400u;
            au[4] = ( w2       & 0x000F000Fu) | 0x64006400u;
            au[5] = ((w2 >> 8) & 0x000F000Fu) | 0x64006400u;
            au[6] = ( w3       & 0x000F000Fu) | 0x64006400u;
            au[7] = ((w3 >> 8) & 0x000F000Fu) | 0x64006400u;
            v16h a = __builtin_bit_cast(v16h, au);
            acc = __builtin_amdgcn_wmma_f32_16x16x32_f16(
                      false, a, false, (ch == 0) ? bfrag0 : bfrag1,
                      (short)0, acc, false, false);
        }
        // C/D layout: lanes 0-15 VGPR v -> M=v; lanes 16-31 -> M=8+v. All N identical.
        if ((lane & 15) == 0) {
            int c = t * 16 + ((lane & 16) ? 8 : 0);
            float4 lo = { acc[0] - corr, acc[1] - corr, acc[2] - corr, acc[3] - corr };
            float4 hi = { acc[4] - corr, acc[5] - corr, acc[6] - corr, acc[7] - corr };
            *(float4*)(out + outBase + c)           = lo;
            *(float4*)(out + outBase + c + 4)       = hi;
            *(float4*)(out + outBase + 128 + c)     = lo;
            *(float4*)(out + outBase + 128 + c + 4) = hi;
        }
    }

    // ---------- phase 5: other_tensor passthrough (coalesced float4) ----------
    const float4* osrc = (const float4*)(other + (size_t)bag * 128);
    float4*       odst = (float4*)(out + outBase + 256);
    odst[lane] = osrc[lane];
}

extern "C" void kernel_launch(void* const* d_in, const int* in_sizes, int n_in,
                              void* d_out, int out_size, void* d_ws, size_t ws_size,
                              hipStream_t stream) {
    const int*   weights = (const int*)  d_in[0];
    const float* scales  = (const float*)d_in[1];
    const float* zeros   = (const float*)d_in[2];
    const int*   indices = (const int*)  d_in[3];
    const int*   offsets = (const int*)  d_in[4];
    const float* other   = (const float*)d_in[5];
    float*       out     = (float*)d_out;

    const int B = in_sizes[4] - 1;   // offsets has B+1 entries
    const int blocks = (B + BAGS_PER_BLOCK - 1) / BAGS_PER_BLOCK;
    quant_embed_bag_wmma<<<blocks, 128, 0, stream>>>(
        weights, scales, zeros, indices, offsets, other, out, B);
}
